// AttentionScanBlock_89799176225254
// MI455X (gfx1250) — compile-verified, matrix operand order checked by reading, not run on hardware
//
#include <hip/hip_runtime.h>
#include <cstdint>
#include <cstddef>

// ---------------------------------------------------------------------------
// Types / helpers
// ---------------------------------------------------------------------------
typedef unsigned short ushort_t;
typedef __attribute__((ext_vector_type(16))) __bf16 v16bf;
typedef __attribute__((ext_vector_type(8)))  float  v8f;

struct U256 { uint4 a, b; };

__device__ __forceinline__ ushort_t f2bf(float f) {
  unsigned u = __float_as_uint(f);
  unsigned r = (u + 0x7FFFu + ((u >> 16) & 1u)) >> 16;   // round-to-nearest-even
  return (ushort_t)r;
}

// Unconditional 16-bit WMMA fragment load from a precomputed per-lane pointer.
// Per ISA layout: lanes 0-15 hold K = [k0..k0+7] and [k0+16..k0+23];
// lanes 16-31 hold K = [k0+8..k0+15] and [k0+24..k0+31].  Caller bakes
// (lane>>4)*8 into the pointer; the two 16B chunks sit at +0 and +16 elems.
__device__ __forceinline__ v16bf load_frag_ptr(const ushort_t* p) {
  U256 u;
  u.a = *reinterpret_cast<const uint4*>(p);
  u.b = *reinterpret_cast<const uint4*>(p + 16);
  return __builtin_bit_cast(v16bf, u);
}

#define WMMA_BF16(A_, B_, C_) \
  __builtin_amdgcn_wmma_f32_16x16x32_bf16(false, (A_), false, (B_), (short)0, (C_), false, false)

// ---------------------------------------------------------------------------
// Elementwise / small kernels
// ---------------------------------------------------------------------------
__global__ void cast_f32_bf16_kernel(const float* __restrict__ in,
                                     ushort_t* __restrict__ out, int n) {
  int i = blockIdx.x * blockDim.x + threadIdx.x;
  if (i < n) out[i] = f2bf(in[i]);
}

__global__ void silu_inplace_kernel(float* __restrict__ g, int n) {
  int i = blockIdx.x * blockDim.x + threadIdx.x;
  if (i < n) { float x = g[i]; g[i] = x / (1.f + __expf(-x)); }
}

__global__ void mul_kernel(const float* __restrict__ a, const float* __restrict__ b,
                           float* __restrict__ out, int n) {
  int i = blockIdx.x * blockDim.x + threadIdx.x;
  if (i < n) out[i] = a[i] * b[i];
}

// One block (256 threads) per row: y = x * rsqrt(mean(x^2)+eps) * w  -> bf16
__global__ __launch_bounds__(256) void rmsnorm_bf16_kernel(
    const float* __restrict__ x, const float* __restrict__ w,
    ushort_t* __restrict__ out, int D) {
  __shared__ float red[256];
  const float* xr = x + (size_t)blockIdx.x * D;
  float ss = 0.f;
  for (int d = threadIdx.x; d < D; d += 256) { float v = xr[d]; ss += v * v; }
  red[threadIdx.x] = ss;
  __syncthreads();
  for (int s = 128; s > 0; s >>= 1) {
    if ((int)threadIdx.x < s) red[threadIdx.x] += red[threadIdx.x + s];
    __syncthreads();
  }
  float inv = rsqrtf(red[0] / (float)D + 1.1920929e-07f);
  ushort_t* orow = out + (size_t)blockIdx.x * D;
  for (int d = threadIdx.x; d < D; d += 256) orow[d] = f2bf(xr[d] * inv * w[d]);
}

// Tiny projections (N<=64, K=64): scalar fp32. act: 0 = none, 1 = softplus(.+bias)
__global__ void small_gemm_kernel(const float* __restrict__ A, const float* __restrict__ W,
                                  const float* __restrict__ bias, float* __restrict__ out,
                                  int M, int N, int K, int act) {
  int i = blockIdx.x * blockDim.x + threadIdx.x;
  if (i >= M * N) return;
  int m = i / N, n = i - m * N;
  const float* a = A + (size_t)m * K;
  const float* w = W + (size_t)n * K;
  float acc = 0.f;
  for (int k = 0; k < K; ++k) acc += a[k] * w[k];
  if (act == 1) {
    acc += bias[n];
    acc = (acc > 20.f) ? acc : log1pf(__expf(acc));
  }
  out[i] = acc;
}

// ---------------------------------------------------------------------------
// Generic bf16 WMMA GEMM:  C[M,N] = A[M,K] @ W[N,K]^T  (+ optional residual)
// Block: 256 threads = 8 waves; wave tile 32x64; block tile 128x128.
// Out-of-range rows/cols are CLAMPED on load (their products land in output
// elements that the guarded store never writes), keeping the K-loop free of
// exec-mask manipulation and zero-fills, and EXEC all-ones for the WMMAs.
// Output: bf16 (Cb != null) or fp32 (Cf).  K must be a multiple of 32.
// ---------------------------------------------------------------------------
__global__ __launch_bounds__(256) void gemm_bf16_wmma_kernel(
    const ushort_t* __restrict__ A, int lda,
    const ushort_t* __restrict__ W, int ldw,
    float* __restrict__ Cf, ushort_t* __restrict__ Cb, int ldc,
    const float* __restrict__ Res,
    int M, int N, int K) {
  const int wave = threadIdx.x >> 5, lane = threadIdx.x & 31;
  const int hl = lane >> 4, l16 = lane & 15;
  const int wr = wave >> 1, wc = wave & 1;
  const int mBase = blockIdx.x * 128 + wr * 32;
  const int nBase = blockIdx.y * 128 + wc * 64;

  // Per-lane fragment base pointers, hoisted out of the K-loop.
  const ushort_t* aPtr[2];
#pragma unroll
  for (int mt = 0; mt < 2; ++mt) {
    int row = mBase + mt * 16 + l16;
    row = (row < M) ? row : (M - 1);
    aPtr[mt] = A + (size_t)row * lda + hl * 8;
  }
  const ushort_t* bPtr[4];
#pragma unroll
  for (int nt = 0; nt < 4; ++nt) {
    int col = nBase + nt * 16 + l16;
    col = (col < N) ? col : (N - 1);
    bPtr[nt] = W + (size_t)col * ldw + hl * 8;
  }

  v8f c[2][4];
#pragma unroll
  for (int mt = 0; mt < 2; ++mt)
#pragma unroll
    for (int nt = 0; nt < 4; ++nt)
      c[mt][nt] = (v8f){0.f, 0.f, 0.f, 0.f, 0.f, 0.f, 0.f, 0.f};

#pragma unroll 2
  for (int k0 = 0; k0 < K; k0 += 32) {
    v16bf a[2], b[4];
#pragma unroll
    for (int mt = 0; mt < 2; ++mt) a[mt] = load_frag_ptr(aPtr[mt] + k0);
#pragma unroll
    for (int nt = 0; nt < 4; ++nt) b[nt] = load_frag_ptr(bPtr[nt] + k0);
#pragma unroll
    for (int mt = 0; mt < 2; ++mt)
#pragma unroll
      for (int nt = 0; nt < 4; ++nt)
        c[mt][nt] = WMMA_BF16(a[mt], b[nt], c[mt][nt]);
  }

  // C layout: lanes 0-15 -> M=vgpr, N=lane; lanes 16-31 -> M=vgpr+8, N=lane-16
#pragma unroll
  for (int mt = 0; mt < 2; ++mt)
#pragma unroll
    for (int nt = 0; nt < 4; ++nt) {
      int col = nBase + nt * 16 + l16;
#pragma unroll
      for (int r = 0; r < 8; ++r) {
        int row = mBase + mt * 16 + hl * 8 + r;
        if (row < M && col < N) {
          float v = c[mt][nt][r];
          if (Res) v += Res[(size_t)row * ldc + col];
          if (Cb) Cb[(size_t)row * ldc + col] = f2bf(v);
          else    Cf[(size_t)row * ldc + col] = v;
        }
      }
    }
}

// ---------------------------------------------------------------------------
// Flash attention (causal), bf16 WMMA for QK^T and P*V, fp32 online softmax.
// qkv: [B*T, 3072] bf16, laid out [q(16 heads x 64) | k | v] per row.
// Grid: (T/128, B*H); 256 threads = 8 waves; each wave owns a 16-row q strip.
// Output: [B*T, 1024] bf16, column = h*64 + d.
// ---------------------------------------------------------------------------
#define ATT_BM 128
#define ATT_BN 128
#define VT_LD  136   // 272B row stride -> 16B aligned, staggered banks

__global__ __launch_bounds__(256) void flash_attn_kernel(
    const ushort_t* __restrict__ qkv, ushort_t* __restrict__ attn_out, int T) {
  __shared__ ushort_t sVt[64 * VT_LD];          // V^T for current key block
  __shared__ ushort_t sP[8 * 16 * ATT_BN];      // per-wave P strips (bf16)

  const int wave = threadIdx.x >> 5, lane = threadIdx.x & 31;
  const int hl = lane >> 4, l16 = lane & 15;
  const int bh = blockIdx.y, b = bh >> 4, h = bh & 15;
  const long rowStride = 3072;

  const ushort_t* qbase = qkv + (size_t)b * T * rowStride + h * 64;
  const ushort_t* kbasep = qbase + 1024;
  const ushort_t* vbasep = qbase + 2048;
  const int qr0 = blockIdx.x * ATT_BM + wave * 16;

  // Q fragments (hd = 64 -> 2 K-steps), kept in registers for whole kernel
  v16bf aq[2];
#pragma unroll
  for (int ks = 0; ks < 2; ++ks)
    aq[ks] = load_frag_ptr(qbase + (size_t)(qr0 + l16) * rowStride + ks * 32 + hl * 8);

  v8f o[4];
#pragma unroll
  for (int j = 0; j < 4; ++j) o[j] = (v8f){0.f, 0.f, 0.f, 0.f, 0.f, 0.f, 0.f, 0.f};
  float rmax[8], rsum[8];
#pragma unroll
  for (int r = 0; r < 8; ++r) { rmax[r] = -3.0e38f; rsum[r] = 0.f; }

  const float scale = 0.125f;  // 1/sqrt(64)
  ushort_t* pP = sP + wave * 16 * ATT_BN;
  // Per-lane K fragment base (key row l16, chunk offset hl*8)
  const ushort_t* kLane = kbasep + (size_t)l16 * rowStride + hl * 8;

  for (int kb = 0; kb <= (int)blockIdx.x; ++kb) {
    const int kbase = kb * ATT_BN;

    __syncthreads();  // protect sVt from previous iteration's readers
    // Stage V^T (128 keys x 64 dims) into LDS as bf16
    for (int idx = threadIdx.x; idx < ATT_BN * 64; idx += 256) {
      int kk = idx >> 6, d = idx & 63;
      sVt[d * VT_LD + kk] = vbasep[(size_t)(kbase + kk) * rowStride + d];
    }
    __syncthreads();

    // S strip: 16 rows x 128 keys, via 8 N-tiles x 2 K-steps of WMMA
    v8f s[8];
#pragma unroll
    for (int nt = 0; nt < 8; ++nt) s[nt] = (v8f){0.f, 0.f, 0.f, 0.f, 0.f, 0.f, 0.f, 0.f};
#pragma unroll
    for (int ks = 0; ks < 2; ++ks) {
#pragma unroll
      for (int nt = 0; nt < 8; ++nt) {
        v16bf bk = load_frag_ptr(kLane + (size_t)(kbase + nt * 16) * rowStride + ks * 32);
        s[nt] = WMMA_BF16(aq[ks], bk, s[nt]);
      }
    }

    // Scale + causal mask (only the diagonal key block needs masking)
    const bool diag = (kb == (int)blockIdx.x);
#pragma unroll
    for (int nt = 0; nt < 8; ++nt) {
      const int kg = kbase + nt * 16 + l16;
#pragma unroll
      for (int r = 0; r < 8; ++r) {
        float v = s[nt][r] * scale;
        if (diag) { int qg = qr0 + hl * 8 + r; if (kg > qg) v = -3.0e38f; }
        s[nt][r] = v;
      }
    }

    // Online softmax per row (row = r + 8*half; reductions stay in 16-lane half)
#pragma unroll
    for (int r = 0; r < 8; ++r) {
      float m = -3.0e38f;
#pragma unroll
      for (int nt = 0; nt < 8; ++nt) m = fmaxf(m, s[nt][r]);
      m = fmaxf(m, __shfl_xor(m, 1, 32));
      m = fmaxf(m, __shfl_xor(m, 2, 32));
      m = fmaxf(m, __shfl_xor(m, 4, 32));
      m = fmaxf(m, __shfl_xor(m, 8, 32));
      float mnew = fmaxf(rmax[r], m);
      float corr = __expf(rmax[r] - mnew);
      rmax[r] = mnew;
      float ls = 0.f;
#pragma unroll
      for (int nt = 0; nt < 8; ++nt) {
        float p = __expf(s[nt][r] - mnew);
        s[nt][r] = p;
        ls += p;
      }
      ls += __shfl_xor(ls, 1, 32);
      ls += __shfl_xor(ls, 2, 32);
      ls += __shfl_xor(ls, 4, 32);
      ls += __shfl_xor(ls, 8, 32);
      rsum[r] = rsum[r] * corr + ls;
#pragma unroll
      for (int j = 0; j < 4; ++j) o[j][r] *= corr;
    }

    // Spill P (bf16) to this wave's LDS strip
#pragma unroll
    for (int nt = 0; nt < 8; ++nt) {
      const int col = nt * 16 + l16;
#pragma unroll
      for (int r = 0; r < 8; ++r)
        pP[(hl * 8 + r) * ATT_BN + col] = f2bf(s[nt][r]);
    }
    __syncthreads();

    // O += P @ V : K = 128 (4 steps), N = 64 (4 tiles)
#pragma unroll
    for (int ks2 = 0; ks2 < 4; ++ks2) {
      const int koff = ks2 * 32 + hl * 8;
      v16bf ap = load_frag_ptr(pP + l16 * ATT_BN + koff);
#pragma unroll
      for (int nt2 = 0; nt2 < 4; ++nt2) {
        v16bf bv = load_frag_ptr(sVt + (nt2 * 16 + l16) * VT_LD + koff);
        o[nt2] = WMMA_BF16(ap, bv, o[nt2]);
      }
    }
  }

  // Epilogue: normalize and write bf16 [B*T, 1024]
#pragma unroll
  for (int nt2 = 0; nt2 < 4; ++nt2) {
    const int col_g = h * 64 + nt2 * 16 + l16;
#pragma unroll
    for (int r = 0; r < 8; ++r) {
      const int row_g = b * T + qr0 + hl * 8 + r;
      attn_out[(size_t)row_g * 1024 + col_g] = f2bf(o[nt2][r] / rsum[r]);
    }
  }
}

// ---------------------------------------------------------------------------
// Selective scan: one block per batch; thread = (sc, st); shuffle-reduce over st.
// ---------------------------------------------------------------------------
__global__ __launch_bounds__(1024) void scan_kernel(
    const float* __restrict__ A_log,  // [64,16]
    const float* __restrict__ dt,     // [B*T,64]
    const float* __restrict__ dtz,    // [B*T,64]
    const float* __restrict__ Bi,     // [B*T,16]
    const float* __restrict__ Ci,     // [B*T,16]
    const float* __restrict__ gate,   // [B*T,64]
    float* __restrict__ y,            // [B*T,64]
    int T) {
  const int bb = blockIdx.x;
  const int sc = threadIdx.x >> 4, st = threadIdx.x & 15;
  const float Aval = -__expf(A_log[sc * 16 + st]);
  float sstate = 0.f;
  for (int t = 0; t < T; ++t) {
    const size_t m = (size_t)bb * T + t;
    const float a_bar = 1.f + dt[m * 64 + sc] * Aval;
    sstate = a_bar * sstate + dtz[m * 64 + sc] * Bi[m * 16 + st];
    float prod = sstate * Ci[m * 16 + st];
    prod += __shfl_xor(prod, 1, 32);
    prod += __shfl_xor(prod, 2, 32);
    prod += __shfl_xor(prod, 4, 32);
    prod += __shfl_xor(prod, 8, 32);
    if (st == 0) y[m * 64 + sc] = prod * gate[m * 64 + sc];
  }
}

// ---------------------------------------------------------------------------
// Host-side orchestration
// ---------------------------------------------------------------------------
static inline void launch_cast(const float* in, ushort_t* out, int n, hipStream_t s) {
  cast_f32_bf16_kernel<<<(n + 255) / 256, 256, 0, s>>>(in, out, n);
}

extern "C" void kernel_launch(void* const* d_in, const int* in_sizes, int n_in,
                              void* d_out, int out_size, void* d_ws, size_t ws_size,
                              hipStream_t stream) {
  (void)in_sizes; (void)n_in; (void)out_size; (void)ws_size;
  const int D = 1024, T = 2048, Bb = 2, SC = 64, ST = 16, Hh = 16;
  const int M = Bb * T;  // 4096 token rows

  const float* x      = (const float*)d_in[0];
  const float* qkv_w  = (const float*)d_in[1];
  const float* o_w    = (const float*)d_in[2];
  const float* n1w    = (const float*)d_in[3];
  const float* n2w    = (const float*)d_in[4];
  const float* in_w   = (const float*)d_in[5];
  const float* out_w  = (const float*)d_in[6];
  const float* A_log  = (const float*)d_in[7];
  const float* Bp_w   = (const float*)d_in[8];
  const float* Cp_w   = (const float*)d_in[9];
  const float* dt_w   = (const float*)d_in[10];
  const float* dt_b   = (const float*)d_in[11];
  const float* gate_w = (const float*)d_in[12];
  float* outp = (float*)d_out;

  // Deterministic workspace carving (256B aligned)
  char* ws = (char*)d_ws;
  size_t off = 0;
  auto alloc = [&](size_t bytes) -> void* {
    void* p = ws + off;
    off = (off + bytes + 255) & ~(size_t)255;
    return p;
  };
  ushort_t* wqkv_bf  = (ushort_t*)alloc((size_t)3 * D * D * 2);
  ushort_t* wo_bf    = (ushort_t*)alloc((size_t)D * D * 2);
  ushort_t* win_bf   = (ushort_t*)alloc((size_t)SC * D * 2);
  ushort_t* wgate_bf = (ushort_t*)alloc((size_t)SC * D * 2);
  ushort_t* wout_bf  = (ushort_t*)alloc((size_t)D * SC * 2);
  ushort_t* h1_bf    = (ushort_t*)alloc((size_t)M * D * 2);
  ushort_t* qkv_bf   = (ushort_t*)alloc((size_t)M * 3 * D * 2);
  ushort_t* attn_bf  = (ushort_t*)alloc((size_t)M * D * 2);
  float*    x2       = (float*)alloc((size_t)M * D * 4);
  ushort_t* h2_bf    = (ushort_t*)alloc((size_t)M * D * 2);
  float*    zf       = (float*)alloc((size_t)M * SC * 4);
  float*    gatef    = (float*)alloc((size_t)M * SC * 4);
  float*    Bif      = (float*)alloc((size_t)M * ST * 4);
  float*    Cif      = (float*)alloc((size_t)M * ST * 4);
  float*    dtf      = (float*)alloc((size_t)M * SC * 4);
  float*    dtzf     = (float*)alloc((size_t)M * SC * 4);
  float*    yf       = (float*)alloc((size_t)M * SC * 4);
  ushort_t* y_bf     = (ushort_t*)alloc((size_t)M * SC * 2);

  // 1) Cast weights to bf16
  launch_cast(qkv_w,  wqkv_bf,  3 * D * D, stream);
  launch_cast(o_w,    wo_bf,    D * D,     stream);
  launch_cast(in_w,   win_bf,   SC * D,    stream);
  launch_cast(gate_w, wgate_bf, SC * D,    stream);
  launch_cast(out_w,  wout_bf,  D * SC,    stream);

  // 2) h1 = rmsnorm(x, norm1_w)  (bf16)
  rmsnorm_bf16_kernel<<<M, 256, 0, stream>>>(x, n1w, h1_bf, D);

  // 3) qkv = h1 @ qkv_w^T  (bf16 out, M=4096 N=3072 K=1024)
  gemm_bf16_wmma_kernel<<<dim3(M / 128, (3 * D) / 128), 256, 0, stream>>>(
      h1_bf, D, wqkv_bf, D, nullptr, qkv_bf, 3 * D, nullptr, M, 3 * D, D);

  // 4) causal flash attention  -> attn_bf [M, D]
  flash_attn_kernel<<<dim3(T / ATT_BM, Bb * Hh), 256, 0, stream>>>(qkv_bf, attn_bf, T);

  // 5) x2 = x + attn @ o_w^T
  gemm_bf16_wmma_kernel<<<dim3(M / 128, D / 128), 256, 0, stream>>>(
      attn_bf, D, wo_bf, D, x2, nullptr, D, x, M, D, D);

  // 6) h2 = rmsnorm(x2, norm2_w)
  rmsnorm_bf16_kernel<<<M, 256, 0, stream>>>(x2, n2w, h2_bf, D);

  // 7) z = h2 @ in_w^T ; gate_pre = h2 @ gate_w^T  (N=64, K=1024)
  gemm_bf16_wmma_kernel<<<dim3(M / 128, 1), 256, 0, stream>>>(
      h2_bf, D, win_bf, D, zf, nullptr, SC, nullptr, M, SC, D);
  gemm_bf16_wmma_kernel<<<dim3(M / 128, 1), 256, 0, stream>>>(
      h2_bf, D, wgate_bf, D, gatef, nullptr, SC, nullptr, M, SC, D);
  silu_inplace_kernel<<<(M * SC + 255) / 256, 256, 0, stream>>>(gatef, M * SC);

  // 8) tiny projections off z (fp32 scalar)
  small_gemm_kernel<<<(M * ST + 255) / 256, 256, 0, stream>>>(zf, Bp_w, nullptr, Bif, M, ST, SC, 0);
  small_gemm_kernel<<<(M * ST + 255) / 256, 256, 0, stream>>>(zf, Cp_w, nullptr, Cif, M, ST, SC, 0);
  small_gemm_kernel<<<(M * SC + 255) / 256, 256, 0, stream>>>(zf, dt_w, dt_b, dtf, M, SC, SC, 1);
  mul_kernel<<<(M * SC + 255) / 256, 256, 0, stream>>>(dtf, zf, dtzf, M * SC);

  // 9) sequential selective scan -> y = scan(...) * gate
  scan_kernel<<<Bb, SC * ST, 0, stream>>>(A_log, dtf, dtzf, Bif, Cif, gatef, yf, T);

  // 10) out = x2 + y @ out_w^T  (M=4096 N=1024 K=64)
  launch_cast(yf, y_bf, M * SC, stream);
  gemm_bf16_wmma_kernel<<<dim3(M / 128, D / 128), 256, 0, stream>>>(
      y_bf, SC, wout_bf, SC, outp, nullptr, D, x2, M, D, SC);
}
